// D_KGTransformer_14628658610678
// MI455X (gfx1250) — matrix operand order repaired and checked by reading, not compiled, not run on hardware
//
#include <hip/hip_runtime.h>

// ---------------------------------------------------------------------------
// Problem constants (match reference)
// ---------------------------------------------------------------------------
#define BB 4
#define SS 256
#define DD 1024
#define HH 16
#define LL 2
#define EE 8
#define DHH 2048
#define DKK 64
#define NTOK (BB*SS)          // 1024
#define NUM_NODES 40000
#define NUM_REL 1000

typedef long long i64;
typedef __attribute__((ext_vector_type(16))) __bf16 v16bf;
typedef __attribute__((ext_vector_type(8)))  __bf16 v8bf;
typedef __attribute__((ext_vector_type(8)))  float  v8f;

// pack two fp32 -> packed bf16x2, round-to-nearest-even (pure integer path)
__device__ __forceinline__ unsigned bfpack(float a, float b) {
    unsigned ua = __float_as_uint(a); ua = ua + 0x7FFFu + ((ua >> 16) & 1u);
    unsigned ub = __float_as_uint(b); ub = ub + 0x7FFFu + ((ub >> 16) & 1u);
    return (ua >> 16) | (ub & 0xFFFF0000u);
}

// ---------------------------------------------------------------------------
// Generic batched GEMM: C = epi( (A@B + biasVec)*alpha + biasMat ), bf16 WMMA
//   batch offset (per tensor): (z>>zsh)*bs_hi + (z&((1<<zsh)-1))*bs_lo
//   A: M x K, element (m,k) at Ab[m*a_rs + k]            (k contiguous)
//   B: K x N, element (k,n) at Bb[k*b_rs + n*b_cs]
// Template:
//   BKCONT : true  -> B k-contiguous (b_rs==1), float4 loads along k
//            false -> B n-contiguous (b_cs==1), 4x4 micro-transpose staging
//   GUARDN : bounds-check N (only needed when N < BN)
// epilogue: val=(acc+biasVec[n])*alpha; val+=biasMat[m,n]; relu; *=rowScale[m];
//           C[m,n] (+)= val
// Block 256 threads = 8 waves (4x2), wave tile 32x64 = 2x4 WMMA accumulators.
// Double-buffered LDS; next K-tile global loads overlap WMMA.
// ---------------------------------------------------------------------------
#define BM 128
#define BN 128
#define BK 32
#define LDSP (BK + 8)   // +8 bf16 pad keeps every fragment 16B aligned

template<bool BKCONT, bool GUARDN>
__global__ __launch_bounds__(256) void gemm_bf16_wmma(
    const float* __restrict__ A, i64 a_rs, i64 a_bhi, i64 a_blo,
    const float* __restrict__ Bm, i64 b_rs, i64 b_cs, i64 b_bhi, i64 b_blo,
    float* __restrict__ C, i64 c_ld, i64 c_bhi, i64 c_blo,
    const float* __restrict__ biasVec,
    const float* __restrict__ biasMat, i64 bm_ld, i64 bm_bhi, i64 bm_blo,
    const float* __restrict__ rowScale,
    float alpha, int relu, int accum, int zsh,
    int M, int N, int K)
{
    __shared__ alignas(16) __bf16 As[2][BM][LDSP];
    __shared__ alignas(16) __bf16 Bs[2][BN][LDSP];   // stored transposed: [n][k]

    const int tid  = threadIdx.x;
    const int lane = tid & 31;
    const int wave = tid >> 5;
    const int wm0  = (wave >> 1) * 32;   // wave row offset in block tile
    const int wn0  = (wave & 1) * 64;    // wave col offset in block tile
    const int hf   = (lane >> 4) & 1;    // half-wave (VGPR layout half)
    const int l16  = lane & 15;

    const int row0 = blockIdx.y * BM;
    const int col0 = blockIdx.x * BN;
    const int zhi  = (int)blockIdx.z >> zsh;
    const int zlo  = (int)blockIdx.z & ((1 << zsh) - 1);

    const float* Ab  = A  + (i64)zhi * a_bhi + (i64)zlo * a_blo;
    const float* Bb  = Bm + (i64)zhi * b_bhi + (i64)zlo * b_blo;
    float*       Cb  = C  + (i64)zhi * c_bhi + (i64)zlo * c_blo;
    const float* BMb = biasMat ? biasMat + (i64)zhi * bm_bhi + (i64)zlo * bm_blo
                               : (const float*)0;

    v8f acc[2][4];
    const v8f zz = {0.f,0.f,0.f,0.f,0.f,0.f,0.f,0.f};
#pragma unroll
    for (int i = 0; i < 2; i++)
#pragma unroll
        for (int j = 0; j < 4; j++) acc[i][j] = zz;

    float4 ra[4];
    float  rbv[4][4];   // !BKCONT: 4x4 micro-tile (k x n)
    float4 rbk[4];      //  BKCONT: float4 along k

    // B staging geometry (!BKCONT): thread owns 4x4 block at (kq, n4)
    const int n4 = (tid & 31) << 2;   // 0..124 (n, coalesced across lanes)
    const int kq = (tid >> 5) << 2;   // 0..28  (k)

    auto ld_tiles = [&](int kk) {
#pragma unroll
        for (int i = 0; i < 4; i++) {               // A: float4 along k
            int g = tid + i * 256;
            int m = g >> 3, k4 = (g & 7) << 2;
            ra[i] = *(const float4*)(Ab + (i64)(row0 + m) * a_rs + (kk + k4));
        }
        if constexpr (BKCONT) {
#pragma unroll
            for (int i = 0; i < 4; i++) {           // B: float4 along k
                int g = tid + i * 256;
                int n = g >> 3, k4 = (g & 7) << 2;
                rbk[i] = *(const float4*)(Bb + (i64)(col0 + n) * b_cs + (kk + k4));
            }
        } else {
            bool ok = !GUARDN || (col0 + n4) < N;
#pragma unroll
            for (int j = 0; j < 4; j++) {           // B: float4 along n, 4 rows
                float4 t = ok ? *(const float4*)(Bb + (i64)(kk + kq + j) * b_rs
                                                 + (col0 + n4))
                              : make_float4(0.f, 0.f, 0.f, 0.f);
                rbv[j][0] = t.x; rbv[j][1] = t.y; rbv[j][2] = t.z; rbv[j][3] = t.w;
            }
        }
    };
    auto st_tiles = [&](int buf) {
#pragma unroll
        for (int i = 0; i < 4; i++) {
            int g = tid + i * 256;
            int m = g >> 3, k4 = (g & 7) << 2;
            uint2 p; p.x = bfpack(ra[i].x, ra[i].y); p.y = bfpack(ra[i].z, ra[i].w);
            *(uint2*)&As[buf][m][k4] = p;
        }
        if constexpr (BKCONT) {
#pragma unroll
            for (int i = 0; i < 4; i++) {
                int g = tid + i * 256;
                int n = g >> 3, k4 = (g & 7) << 2;
                uint2 p; p.x = bfpack(rbk[i].x, rbk[i].y); p.y = bfpack(rbk[i].z, rbk[i].w);
                *(uint2*)&Bs[buf][n][k4] = p;
            }
        } else {                                     // register 4x4 transpose
#pragma unroll
            for (int j = 0; j < 4; j++) {
                uint2 p;
                p.x = bfpack(rbv[0][j], rbv[1][j]);
                p.y = bfpack(rbv[2][j], rbv[3][j]);
                *(uint2*)&Bs[buf][n4 + j][kq] = p;
            }
        }
    };

    ld_tiles(0);
    st_tiles(0);
    int buf = 0;

    for (int k0 = 0; k0 < K; k0 += BK) {
        __syncthreads();
        const bool more = (k0 + BK) < K;
        if (more) ld_tiles(k0 + BK);    // overlap next-tile globals with WMMA

        // ---- build WMMA fragments from LDS (16B ds_read chunks) ----
        // A 16x32 bf16 layout: elems 0..7 -> K = hf*8 + e ; 8..15 -> 16 + hf*8 + e
        v16bf af[2];
#pragma unroll
        for (int wm = 0; wm < 2; wm++) {
            int r = wm0 + wm * 16 + l16;
            v8bf lo = *(const v8bf*)&As[buf][r][hf * 8];
            v8bf hi = *(const v8bf*)&As[buf][r][16 + hf * 8];
#pragma unroll
            for (int e = 0; e < 8; e++) { af[wm][e] = lo[e]; af[wm][8 + e] = hi[e]; }
        }
        // B 32x16 bf16 layout: elems 0..7 -> K = hf*8 + e ; 8..15 -> 16 + hf*8 + e
        v16bf bfr[4];
#pragma unroll
        for (int wn = 0; wn < 4; wn++) {
            int c = wn0 + wn * 16 + l16;
            v8bf lo = *(const v8bf*)&Bs[buf][c][hf * 8];
            v8bf hi = *(const v8bf*)&Bs[buf][c][16 + hf * 8];
#pragma unroll
            for (int e = 0; e < 8; e++) { bfr[wn][e] = lo[e]; bfr[wn][8 + e] = hi[e]; }
        }

#pragma unroll
        for (int wm = 0; wm < 2; wm++)
#pragma unroll
            for (int wn = 0; wn < 4; wn++)
                acc[wm][wn] = __builtin_amdgcn_wmma_f32_16x16x32_bf16(
                    false, af[wm], false, bfr[wn], (short)0, acc[wm][wn],
                    false, false);

        if (more) st_tiles(buf ^ 1);    // other buffer: no barrier needed here
        buf ^= 1;
    }

    // ---- epilogue; C/D layout: m = v + hf*8, n = l16 within each 16x16 tile ----
#pragma unroll
    for (int wm = 0; wm < 2; wm++)
#pragma unroll
        for (int wn = 0; wn < 4; wn++) {
            int gc = col0 + wn0 + wn * 16 + l16;
            if (GUARDN && gc >= N) continue;
#pragma unroll
            for (int v = 0; v < 8; v++) {
                int gr = row0 + wm0 + wm * 16 + hf * 8 + v;
                float val = acc[wm][wn][v];
                val = (val + (biasVec ? biasVec[gc] : 0.f)) * alpha;
                if (BMb) val += BMb[(i64)gr * bm_ld + gc];
                if (relu) val = fmaxf(val, 0.f);
                if (rowScale) val *= rowScale[gr];
                i64 off = (i64)gr * c_ld + gc;
                if (accum) Cb[off] += val; else Cb[off] = val;
            }
        }
}

// ---------------------------------------------------------------------------
// Embedding gather: feat[tok,d] = type_emb[t] + {ent|mask|rel}
// ---------------------------------------------------------------------------
__global__ void embed_kernel(const int* __restrict__ et, const int* __restrict__ xi,
                             const float* __restrict__ ent, const float* __restrict__ msk,
                             const float* __restrict__ rel, const float* __restrict__ typ,
                             float* __restrict__ feat)
{
    int idx = blockIdx.x * blockDim.x + threadIdx.x;   // NTOK*DD
    int tok = idx >> 10, d = idx & 1023;
    int t = et[tok];
    const float* src;
    if (t == 0)      src = ent + (i64)(xi[tok] % NUM_NODES) * DD;
    else if (t == 1) src = msk;
    else             src = rel + (i64)(xi[tok] % NUM_REL) * DD;
    feat[idx] = typ[t * DD + d] + src[d];
}

// bias[b,h,i,j] = bias_emb[attn_bias_type[b,i,j], h]
__global__ void bias_kernel(const int* __restrict__ abt, const float* __restrict__ be,
                            float* __restrict__ bias)
{
    int idx = blockIdx.x * blockDim.x + threadIdx.x;   // B*H*S*S
    int j = idx & 255, i = (idx >> 8) & 255, h = (idx >> 16) & 15, b = idx >> 20;
    int t = abt[(b << 16) + (i << 8) + j];
    bias[idx] = be[t * HH + h];
}

// ---------------------------------------------------------------------------
// LayerNorm, wave(32)-per-row, 8 rows per 256-thread block
// ---------------------------------------------------------------------------
__global__ __launch_bounds__(256) void ln_kernel(const float* __restrict__ x,
                                                 const float* __restrict__ g,
                                                 const float* __restrict__ b,
                                                 float* __restrict__ y)
{
    int wave = threadIdx.x >> 5, lane = threadIdx.x & 31;
    i64 row = (i64)blockIdx.x * 8 + wave;
    const float* xr = x + row * DD;
    float s = 0.f;
    for (int d = lane; d < DD; d += 32) s += xr[d];
#pragma unroll
    for (int o = 16; o; o >>= 1) s += __shfl_xor(s, o, 32);
    float mean = s * (1.f / DD);
    float var = 0.f;
    for (int d = lane; d < DD; d += 32) { float t = xr[d] - mean; var += t * t; }
#pragma unroll
    for (int o = 16; o; o >>= 1) var += __shfl_xor(var, o, 32);
    float rstd = rsqrtf(var * (1.f / DD) + 1e-5f);
    float* yr = y + row * DD;
    for (int d = lane; d < DD; d += 32)
        yr[d] = (xr[d] - mean) * rstd * g[d] + b[d];
}

// softmax over rows of length 256 (wave per row, 8 elems/lane)
__global__ __launch_bounds__(256) void softmax_kernel(float* __restrict__ p)
{
    int wave = threadIdx.x >> 5, lane = threadIdx.x & 31;
    i64 row = (i64)blockIdx.x * 8 + wave;
    float* pr = p + row * SS;
    float v[8], mx = -3.4e38f;
#pragma unroll
    for (int i = 0; i < 8; i++) { v[i] = pr[lane + i * 32]; mx = fmaxf(mx, v[i]); }
#pragma unroll
    for (int o = 16; o; o >>= 1) mx = fmaxf(mx, __shfl_xor(mx, o, 32));
    float sum = 0.f;
#pragma unroll
    for (int i = 0; i < 8; i++) { v[i] = __expf(v[i] - mx); sum += v[i]; }
#pragma unroll
    for (int o = 16; o; o >>= 1) sum += __shfl_xor(sum, o, 32);
    float inv = 1.f / sum;
#pragma unroll
    for (int i = 0; i < 8; i++) pr[lane + i * 32] = v[i] * inv;
}

// ---------------------------------------------------------------------------
// Top-2 gate: block per token; wave e computes logit_e; thread0 does top-2
// Writes routing weights transposed: r_t[e*NTOK + token]
// ---------------------------------------------------------------------------
__global__ __launch_bounds__(256) void gate_kernel(const float* __restrict__ z,
                                                   const float* __restrict__ gw,
                                                   const float* __restrict__ gb,
                                                   float* __restrict__ r_t)
{
    __shared__ float logit[EE];
    int tok = blockIdx.x, wave = threadIdx.x >> 5, lane = threadIdx.x & 31;
    const float* zr = z + (i64)tok * DD;
    float acc = 0.f;
    for (int d = lane; d < DD; d += 32) acc += zr[d] * gw[d * EE + wave];
#pragma unroll
    for (int o = 16; o; o >>= 1) acc += __shfl_xor(acc, o, 32);
    if (lane == 0) logit[wave] = acc + gb[wave];
    __syncthreads();
    if (threadIdx.x == 0) {
        int i1 = 0;
        for (int e = 1; e < EE; e++) if (logit[e] > logit[i1]) i1 = e;
        int i2 = (i1 == 0) ? 1 : 0;
        for (int e = 0; e < EE; e++) { if (e == i1) continue; if (logit[e] > logit[i2]) i2 = e; }
        float e2 = __expf(logit[i2] - logit[i1]);
        float inv = 1.f / (1.f + e2);
        for (int e = 0; e < EE; e++) r_t[e * NTOK + tok] = 0.f;
        r_t[i1 * NTOK + tok] = inv;
        r_t[i2 * NTOK + tok] = e2 * inv;
    }
}

// ---------------------------------------------------------------------------
// Host orchestration
// ---------------------------------------------------------------------------
extern "C" void kernel_launch(void* const* d_in, const int* in_sizes, int n_in,
                              void* d_out, int out_size, void* d_ws, size_t ws_size,
                              hipStream_t stream)
{
    (void)in_sizes; (void)n_in; (void)out_size; (void)ws_size;
    const int*   embed_type = (const int*)d_in[0];
    const int*   xi         = (const int*)d_in[1];
    const int*   abt        = (const int*)d_in[2];
    const float* ent_emb    = (const float*)d_in[3];
    const float* mask_emb   = (const float*)d_in[4];
    const float* rel_emb    = (const float*)d_in[5];
    const float* type_emb   = (const float*)d_in[6];
    const float* bias_emb   = (const float*)d_in[7];
    const float* ln1_g = (const float*)d_in[8],  *ln1_b = (const float*)d_in[9];
    const float* wq = (const float*)d_in[10], *bq = (const float*)d_in[11];
    const float* wk = (const float*)d_in[12], *bk = (const float*)d_in[13];
    const float* wv = (const float*)d_in[14], *bv = (const float*)d_in[15];
    const float* wo = (const float*)d_in[16], *bo = (const float*)d_in[17];
    const float* ln2_g = (const float*)d_in[18], *ln2_b = (const float*)d_in[19];
    const float* gate_w = (const float*)d_in[20], *gate_b = (const float*)d_in[21];
    const float* w1 = (const float*)d_in[22], *b1 = (const float*)d_in[23];
    const float* w2 = (const float*)d_in[24], *b2 = (const float*)d_in[25];
    const float* fln_g = (const float*)d_in[26], *fln_b = (const float*)d_in[27];
    float* out = (float*)d_out;

    // workspace carve (floats)
    float* feat  = (float*)d_ws;            // 1M  [NTOK, D]
    float* ybuf  = feat  + 1048576;         // 1M  ln out / attn rows [B,S,H*DK]
    float* qful  = ybuf  + 1048576;         // 1M  [B,S,H,DK]
    float* kful  = qful  + 1048576;         // 1M
    float* vful  = kful  + 1048576;         // 1M
    float* probs = vful  + 1048576;         // 4M  [B,H,S,S]
    float* biasb = probs + 4194304;         // 4M  [B,H,S,S]
    float* hbuf  = biasb + 4194304;         // 2M  [NTOK, DH]
    float* rt    = hbuf  + 2097152;         // 8K  [E, NTOK]

    const dim3 blk(256);
    const i64 SD  = (i64)SS * DD;           // 262144 : per-b stride in [B,S,D]
    const i64 S2  = (i64)SS * SS;           // 65536  : per-head stride in [B,H,S,S]
    const i64 S2B = S2 * HH;                // 1048576: per-b stride in [B,H,S,S]

    embed_kernel<<<4096, blk, 0, stream>>>(embed_type, xi, ent_emb, mask_emb,
                                           rel_emb, type_emb, feat);
    bias_kernel<<<16384, blk, 0, stream>>>(abt, bias_emb, biasb);

    for (int l = 0; l < LL; l++) {
        const i64 lDD = (i64)l * DD * DD;
        // ---- LN1 ----
        ln_kernel<<<128, blk, 0, stream>>>(feat, ln1_g + l * DD, ln1_b + l * DD, ybuf);
        // ---- Q/K/V projections (Q folds 1/sqrt(DK)); stay in [B,S,H*DK] ----
        gemm_bf16_wmma<false,false><<<dim3(8, 8, 1), blk, 0, stream>>>(
            ybuf, DD, 0, 0, wq + lDD, DD, 1, 0, 0, qful, DD, 0, 0,
            bq + l * DD, nullptr, 0, 0, 0, nullptr, 0.125f, 0, 0, 0, NTOK, DD, DD);
        gemm_bf16_wmma<false,false><<<dim3(8, 8, 1), blk, 0, stream>>>(
            ybuf, DD, 0, 0, wk + lDD, DD, 1, 0, 0, kful, DD, 0, 0,
            bk + l * DD, nullptr, 0, 0, 0, nullptr, 1.0f, 0, 0, 0, NTOK, DD, DD);
        gemm_bf16_wmma<false,false><<<dim3(8, 8, 1), blk, 0, stream>>>(
            ybuf, DD, 0, 0, wv + lDD, DD, 1, 0, 0, vful, DD, 0, 0,
            bv + l * DD, nullptr, 0, 0, 0, nullptr, 1.0f, 0, 0, 0, NTOK, DD, DD);
        // ---- scores = Q @ K^T + bias, z = b*16+h (zsh=4), direct indexing ----
        gemm_bf16_wmma<true,false><<<dim3(2, 2, BB * HH), blk, 0, stream>>>(
            qful, DD, SD, DKK,                   // A: q[b, s, h*64 + k]
            kful, 1, DD, SD, DKK,                // B: k[b, n, h*64 + kdim]
            probs, SS, S2B, S2,                  // C: probs[b, h, m, n]
            nullptr, biasb, SS, S2B, S2,         // + attn bias matrix
            nullptr, 1.0f, 0, 0, 4, SS, SS, DKK);
        softmax_kernel<<<2048, blk, 0, stream>>>(probs);
        // ---- O = P @ V (N=64 -> guarded); write straight to [B,S,H*DK] ----
        gemm_bf16_wmma<false,true><<<dim3(1, 2, BB * HH), blk, 0, stream>>>(
            probs, SS, S2B, S2,                  // A: probs[b, h, m, k]
            vful, DD, 1, SD, DKK,                // B: v[b, k, h*64 + n]
            ybuf, DD, SD, DKK,                   // C: y[b, m, h*64 + n]
            nullptr, nullptr, 0, 0, 0,
            nullptr, 1.0f, 0, 0, 4, SS, DKK, SS);
        // ---- output projection, residual accumulate into feat ----
        gemm_bf16_wmma<false,false><<<dim3(8, 8, 1), blk, 0, stream>>>(
            ybuf, DD, 0, 0, wo + lDD, DD, 1, 0, 0, feat, DD, 0, 0,
            bo + l * DD, nullptr, 0, 0, 0, nullptr, 1.0f, 0, 1, 0, NTOK, DD, DD);
        // ---- LN2 + MoE ----
        ln_kernel<<<128, blk, 0, stream>>>(feat, ln2_g + l * DD, ln2_b + l * DD, ybuf);
        gate_kernel<<<NTOK, blk, 0, stream>>>(ybuf, gate_w + (i64)l * DD * EE,
                                              gate_b + l * EE, rt);
        for (int e = 0; e < EE; e++) {
            const i64 we = (i64)(l * EE + e);
            gemm_bf16_wmma<false,false><<<dim3(16, 8, 1), blk, 0, stream>>>(
                ybuf, DD, 0, 0, w1 + we * DD * DHH, DHH, 1, 0, 0, hbuf, DHH, 0, 0,
                b1 + we * DHH, nullptr, 0, 0, 0, nullptr, 1.0f, /*relu*/1, 0, 0,
                NTOK, DHH, DD);
            gemm_bf16_wmma<false,false><<<dim3(8, 8, 1), blk, 0, stream>>>(
                hbuf, DHH, 0, 0, w2 + we * DHH * DD, DD, 1, 0, 0, feat, DD, 0, 0,
                b2 + we * DD, nullptr, 0, 0, 0, rt + e * NTOK, 1.0f, 0, /*accum*/1, 0,
                NTOK, DD, DHH);
        }
    }
    // ---- final LN straight into d_out ----
    ln_kernel<<<128, blk, 0, stream>>>(feat, fln_g, fln_b, out);
}
